// Head_63187558858829
// MI455X (gfx1250) — matrix operand order repaired and checked by reading, not compiled, not run on hardware
//
#include <hip/hip_runtime.h>

typedef __attribute__((ext_vector_type(16))) __bf16 v16bf;
typedef __attribute__((ext_vector_type(8)))  __bf16 v8bf;
typedef __attribute__((ext_vector_type(8)))  float  v8f;

#define T_CTX 100
#define C_DIM 384
#define H_DIM 64
#define TP 112      // T padded to 7 tiles of 16
#define XP 392      // xb row pitch (bf16), mult of 8
#define QP 72       // q/k row pitch (bf16)
#define VP 128      // vT row pitch (bf16), s padded to 128
#define SP 120      // scores row pitch (f32)
#define PP 128      // probs row pitch (bf16), K padded to 128

// LDS byte offsets (region A reused: xb -> S + P)
#define S_OFF 0
#define P_OFF 53760            // 112*120*4
#define Q_OFF 87808            // 112*392*2 (xb size)
#define K_OFF (Q_OFF + 16128)  // 112*72*2
#define V_OFF (K_OFF + 16128)
#define SMEM_BYTES (V_OFF + 16384)  // + 64*128*2 = 136448

union AB { v16bf v; v8bf h[2]; };
union F8 { v8f v; float f[8]; };

// A fragment: 16x32 bf16, lanes 0-15 hold row m, K {kb..kb+7, 16+kb..16+kb+7}, kb = (lane/16)*8
__device__ __forceinline__ v16bf load_a(const __bf16* base, int pitch, int m0, int c0, int lane) {
    int m = m0 + (lane & 15);
    int kb = (lane >> 4) << 3;
    const __bf16* p = base + m * pitch + c0 + kb;
    AB u;
    u.h[0] = *(const v8bf*)(p);
    u.h[1] = *(const v8bf*)(p + 16);
    return u.v;
}
// B fragment: 32x16 bf16 from a K-contiguous (transposed) operand BT[n][k]:
// lane holds column n, K {kb..kb+7, 16+kb..16+kb+7}, kb = (lane/16)*8
__device__ __forceinline__ v16bf load_b(const __bf16* base, int pitch, int n0, int c0, int lane) {
    int n = n0 + (lane & 15);
    int kb = (lane >> 4) << 3;
    const __bf16* p = base + n * pitch + c0 + kb;
    AB u;
    u.h[0] = *(const v8bf*)(p);
    u.h[1] = *(const v8bf*)(p + 16);
    return u.v;
}

__global__ void prep_w(const float* __restrict__ Wq, const float* __restrict__ Wk,
                       const float* __restrict__ Wv, __bf16* __restrict__ WT) {
    int idx = blockIdx.x * 256 + threadIdx.x;
    if (idx >= 3 * C_DIM * H_DIM) return;
    int w = idx / (C_DIM * H_DIM);
    int rem = idx - w * (C_DIM * H_DIM);
    int c = rem / H_DIM;
    int h = rem - c * H_DIM;
    const float* W = (w == 0) ? Wq : ((w == 1) ? Wk : Wv);
    WT[w * H_DIM * C_DIM + h * C_DIM + c] = (__bf16)W[c * H_DIM + h];
}

__global__ __launch_bounds__(256, 2)
void attn_head(const float* __restrict__ x, const __bf16* __restrict__ WT,
               float* __restrict__ out) {
    __shared__ __align__(16) unsigned char smem[SMEM_BYTES];
    __bf16* xb = (__bf16*)smem;                 // [112][392] bf16   (phase 1-2)
    float*  Sm = (float*)(smem + S_OFF);        // [112][120] f32    (phase 3-4, aliases xb)
    __bf16* Pm = (__bf16*)(smem + P_OFF);       // [112][128] bf16   (phase 4-5, aliases xb)
    __bf16* qs = (__bf16*)(smem + Q_OFF);       // [112][72]  bf16
    __bf16* ks = (__bf16*)(smem + K_OFF);       // [112][72]  bf16
    __bf16* vT = (__bf16*)(smem + V_OFF);       // [64][128]  bf16 (transposed v)

    const int tid  = threadIdx.x;
    const int lane = tid & 31;
    const int wave = tid >> 5;
    const int b    = blockIdx.x;
    const float* xg = x + (size_t)b * (T_CTX * C_DIM);

    // ---- Phase 1: x fp32 -> xb bf16, zero-pad rows [100,112) ----
    for (int p = tid; p < TP * (C_DIM / 2); p += 256) {
        int row = p / (C_DIM / 2);
        int c2  = (p - row * (C_DIM / 2)) * 2;
        float a0 = 0.f, a1 = 0.f;
        if (row < T_CTX) {
            float2 t = *(const float2*)(xg + row * C_DIM + c2);
            a0 = t.x; a1 = t.y;
        }
        xb[row * XP + c2]     = (__bf16)a0;
        xb[row * XP + c2 + 1] = (__bf16)a1;
    }
    __syncthreads();

    // ---- Phase 2: q,k,v = xb * W   (3 * 7 * 4 = 84 tiles, K = 384) ----
    for (int tile = wave; tile < 84; tile += 8) {
        int which = tile / 28;
        int rem = tile - which * 28;
        int mt = rem >> 2, nt = rem & 3;
        const __bf16* Wg = WT + which * (H_DIM * C_DIM);
        F8 acc;
        for (int i = 0; i < 8; ++i) acc.f[i] = 0.f;
        for (int kc = 0; kc < 12; ++kc) {
            v16bf a  = load_a(xb, XP, mt * 16, kc * 32, lane);
            v16bf bb = load_b(Wg, C_DIM, nt * 16, kc * 32, lane);
            acc.v = __builtin_amdgcn_wmma_f32_16x16x32_bf16(false, a, false, bb,
                                                            (short)0, acc.v, false, false);
        }
        int row0 = mt * 16 + ((lane >> 4) << 3);
        int col  = nt * 16 + (lane & 15);
        if (which == 0) {
            for (int r = 0; r < 8; ++r) qs[(row0 + r) * QP + col] = (__bf16)acc.f[r];
        } else if (which == 1) {
            for (int r = 0; r < 8; ++r) ks[(row0 + r) * QP + col] = (__bf16)acc.f[r];
        } else {
            for (int r = 0; r < 8; ++r) vT[col * VP + (row0 + r)] = (__bf16)acc.f[r];  // transposed
        }
    }
    // zero-pad vT columns s in [112,128)
    for (int i = tid; i < H_DIM * 16; i += 256) {
        int h = i >> 4;
        vT[h * VP + TP + (i & 15)] = (__bf16)0.f;
    }
    __syncthreads();

    // ---- Phase 3: S = (q k^T) * sqrt(64), causal mask   (49 tiles, K = 64) ----
    for (int tile = wave; tile < 49; tile += 8) {
        int mt = tile / 7, nt = tile - mt * 7;
        F8 acc;
        for (int i = 0; i < 8; ++i) acc.f[i] = 0.f;
        for (int kc = 0; kc < 2; ++kc) {
            v16bf a  = load_a(qs, QP, mt * 16, kc * 32, lane);
            v16bf bb = load_b(ks, QP, nt * 16, kc * 32, lane);  // k[s][h] is already BT for q.kT
            acc.v = __builtin_amdgcn_wmma_f32_16x16x32_bf16(false, a, false, bb,
                                                            (short)0, acc.v, false, false);
        }
        int col  = nt * 16 + (lane & 15);
        int row0 = mt * 16 + ((lane >> 4) << 3);
        for (int r = 0; r < 8; ++r) {
            int row = row0 + r;
            float s = acc.f[r] * 8.0f;           // faithful to source bug: *sqrt(H)
            if (col > row) s = -__builtin_inff();
            Sm[row * SP + col] = s;
        }
    }
    __syncthreads();

    // ---- Phase 4: row softmax -> P bf16, zero-pad K to 128 ----
    for (int r = tid; r < TP; r += 256) {
        const float* srow = Sm + r * SP;
        float m = -__builtin_inff();
        for (int c = 0; c < TP; ++c) m = fmaxf(m, srow[c]);
        float sum = 0.f;
        for (int c = 0; c < TP; ++c) sum += __expf(srow[c] - m);
        float inv = 1.0f / sum;
        __bf16* prow = Pm + r * PP;
        for (int c = 0; c < TP; ++c) prow[c] = (__bf16)(__expf(srow[c] - m) * inv);
        for (int c = TP; c < PP; ++c) prow[c] = (__bf16)0.f;
    }
    __syncthreads();

    // ---- Phase 5: O = P * v   (28 tiles, K = 128 padded) ----
    float* og = out + (size_t)b * (T_CTX * H_DIM);
    for (int tile = wave; tile < 28; tile += 8) {
        int mt = tile >> 2, nt = tile & 3;
        F8 acc;
        for (int i = 0; i < 8; ++i) acc.f[i] = 0.f;
        for (int kc = 0; kc < 4; ++kc) {
            v16bf a  = load_a(Pm, PP, mt * 16, kc * 32, lane);
            v16bf bb = load_b(vT, VP, nt * 16, kc * 32, lane);
            acc.v = __builtin_amdgcn_wmma_f32_16x16x32_bf16(false, a, false, bb,
                                                            (short)0, acc.v, false, false);
        }
        int col  = nt * 16 + (lane & 15);
        int row0 = mt * 16 + ((lane >> 4) << 3);
        for (int r = 0; r < 8; ++r) {
            int row = row0 + r;
            if (row < T_CTX) og[row * H_DIM + col] = acc.f[r];
        }
    }
}

extern "C" void kernel_launch(void* const* d_in, const int* in_sizes, int n_in,
                              void* d_out, int out_size, void* d_ws, size_t ws_size,
                              hipStream_t stream) {
    const float* x  = (const float*)d_in[0];
    const float* Wq = (const float*)d_in[1];
    const float* Wk = (const float*)d_in[2];
    const float* Wv = (const float*)d_in[3];
    __bf16* WT = (__bf16*)d_ws;   // 3 * 64 * 384 bf16 = 147456 bytes
    int B = in_sizes[0] / (T_CTX * C_DIM);

    int nprep = 3 * C_DIM * H_DIM;
    prep_w<<<(nprep + 255) / 256, 256, 0, stream>>>(Wq, Wk, Wv, WT);
    attn_head<<<B, 256, 0, stream>>>(x, WT, (float*)d_out);
}
